// InvariantPointAttentionV43_4861902979614
// MI455X (gfx1250) — matrix operand order repaired and checked by reading, not compiled, not run on hardware
//
#include <hip/hip_runtime.h>
#include <hip/hip_bf16.h>

typedef __attribute__((ext_vector_type(16))) _Float16 v16h;
typedef __attribute__((ext_vector_type(8)))  _Float16 v8h;
typedef __attribute__((ext_vector_type(8)))  float    v8f;
typedef __attribute__((ext_vector_type(4)))  float    v4f;

#define IPA_B  2
#define IPA_N  512
#define IPA_C  384
#define IPA_H  12
#define IPA_P  4
#define IPA_DP 128
#define IPA_CH 32
#define IPA_KE 64                       // 32 scalar + 12 point + 20 zero pad (K%32==0)
#define IPA_KPO 160                     // Wpo K padded 144 -> 160
#define IPA_SCALE 0.17677669529663687f  // 1/sqrt(32)
#define IPA_EPS 1e-5f

// ---------------------------------------------------------------------------
// Tile-exact WMMA GEMM:  C = alpha * (A f32->f16) x B + bias + addend
//   A : f32 [M,K] row-major (lda)           -- M%16==0, K%32==0
//   Bt: f16 [N,K] row-major (ldb), i.e. column-major B; k contiguous per column
//   C : f32 [M,N] row-major (ldc)           -- N%16==0
// One wave computes one 16x16 tile via v_wmma_f32_16x16x32_f16.
// Fragment layouts per CDNA5 ISA 7.12.2:
//   A 16x32 f16: lanes 0-15 row=lane, k runs [kk..kk+7],[kk+16..kk+23];
//                lanes 16-31 row=lane-16, k runs [kk+8..+15],[kk+24..+31]
//   B 32x16 f16: same k runs as A; lanes 0-15 col=lane, lanes 16-31 col=lane-16
//   C f32: VGPR i -> row 16*tm+i (+8 for lanes 16-31), col = 16*tn + lane%16
// All loads are contiguous 16-byte vectors; no bounds checks anywhere.
// ---------------------------------------------------------------------------
__global__ __launch_bounds__(256)
void ipa_gemm_wmma(const float* __restrict__ A,
                   const _Float16* __restrict__ Bt,
                   const float* __restrict__ bias,
                   const float* __restrict__ addend,
                   float* __restrict__ Cm,
                   int M, int Nn, int K,
                   int lda, int ldb, int ldc,
                   long long sA, long long sB, long long sC,
                   float alpha)
{
    const int batch = blockIdx.y;
    A  += (size_t)batch * (size_t)sA;
    Bt += (size_t)batch * (size_t)sB;
    Cm += (size_t)batch * (size_t)sC;
    if (addend) addend += (size_t)batch * (size_t)sC;

    const int tilesN = Nn >> 4;
    const int tilesM = M >> 4;
    const int total  = tilesM * tilesN;
    const int wave   = threadIdx.x >> 5;
    const int tile   = blockIdx.x * 8 + wave;
    if (tile >= total) return;            // uniform within the wave
    const int tm   = tile / tilesN;
    const int tn   = tile % tilesN;
    const int lane = threadIdx.x & 31;
    const int lh   = lane & 15;
    const bool hiL = lane >= 16;

    const float*    ap = A  + (size_t)(tm * 16 + lh) * lda + (hiL ? 8 : 0);
    const _Float16* bp = Bt + (size_t)(tn * 16 + lh) * ldb + (hiL ? 8 : 0);

    v8f acc = {};
    #pragma unroll 2
    for (int kk = 0; kk < K; kk += 32) {
        v4f a0 = *(const v4f*)(ap);
        v4f a1 = *(const v4f*)(ap + 4);
        v4f a2 = *(const v4f*)(ap + 16);
        v4f a3 = *(const v4f*)(ap + 20);
        v8h b0 = *(const v8h*)(bp);
        v8h b1 = *(const v8h*)(bp + 16);
        __builtin_prefetch(ap + 32, 0, 0);      // speculative; safe past end
        v16h af, bf;
        #pragma unroll
        for (int e = 0; e < 4; ++e) {
            af[e]      = (_Float16)a0[e];
            af[4 + e]  = (_Float16)a1[e];
            af[8 + e]  = (_Float16)a2[e];
            af[12 + e] = (_Float16)a3[e];
        }
        #pragma unroll
        for (int e = 0; e < 8; ++e) {
            bf[e]     = b0[e];
            bf[8 + e] = b1[e];
        }
        acc = __builtin_amdgcn_wmma_f32_16x16x32_f16(false, af, false, bf,
                                                     (short)0, acc, false, false);
        ap += 32;
        bp += 32;
    }

    const int cn = tn * 16 + lh;
    const float bv = bias ? bias[cn] : 0.0f;
    #pragma unroll
    for (int i = 0; i < 8; ++i) {
        int row = tm * 16 + i + (hiL ? 8 : 0);
        size_t idx = (size_t)row * ldc + cn;
        float o = alpha * acc[i] + bv;
        if (addend) o += addend[idx];
        Cm[idx] = o;
    }
}

// ---------------------------------------------------------------------------
// Pack weight W f32 [K,N] -> Bt f16 [Npad,Kpad] (column-major B, zero padded),
// optionally pad bias to bpad[Npad].
// ---------------------------------------------------------------------------
__global__ void ipa_pack_bt(const float* __restrict__ W,
                            const float* __restrict__ bsrc,
                            _Float16* __restrict__ Wt,
                            float* __restrict__ bpad,
                            int K, int N, int Kpad, int Npad)
{
    int idx = blockIdx.x * blockDim.x + threadIdx.x;
    if (idx >= Kpad * Npad) return;
    int n = idx / Kpad;
    int k = idx % Kpad;
    float x = (k < K && n < N) ? W[(size_t)k * N + n] : 0.0f;
    Wt[(size_t)n * Kpad + k] = (_Float16)x;
    if (bpad && k == 0) bpad[n] = (n < N) ? bsrc[n] : 0.0f;
}

// ---------------------------------------------------------------------------
// In-place frame transform: pts[b,n,h,p,:] = pts @ rot[b,n] + trans[b,n]
// ---------------------------------------------------------------------------
__global__ void ipa_point_to_global(float* __restrict__ pts,
                                    const float* __restrict__ rot,
                                    const float* __restrict__ trans)
{
    int idx = blockIdx.x * blockDim.x + threadIdx.x;
    if (idx >= IPA_B * IPA_N * IPA_H * IPA_P) return;
    int bn = idx / (IPA_H * IPA_P);
    int hp = idx % (IPA_H * IPA_P);
    float* p = pts + (size_t)bn * (IPA_H * IPA_P * 3) + hp * 3;
    const float* R = rot   + (size_t)bn * 9;
    const float* t = trans + (size_t)bn * 3;
    float x = p[0], y = p[1], z = p[2];
    float gx = x * R[0] + y * R[3] + z * R[6] + t[0];
    float gy = x * R[1] + y * R[4] + z * R[7] + t[1];
    float gz = x * R[2] + y * R[5] + z * R[8] + t[2];
    p[0] = gx; p[1] = gy; p[2] = gz;
}

// ---------------------------------------------------------------------------
// Extended feature rows.  qe (f32, A operand): [B*H*N, 64].
// ke (f16, Bt operand for logits GEMM): [B*H*N, 64].
// veT (f16, Bt operand for apply GEMM): [B*H, 64, 512] (component-major).
// sq[b,n,h] = sum of squared global points.
// ---------------------------------------------------------------------------
__global__ void ipa_build_q(const float* __restrict__ sc, const float* __restrict__ pg,
                            float* __restrict__ ext, float* __restrict__ sq)
{
    int idx = blockIdx.x * blockDim.x + threadIdx.x;
    if (idx >= IPA_B * IPA_N * IPA_H) return;
    int b = idx / (IPA_N * IPA_H);
    int n = (idx / IPA_H) % IPA_N;
    int h = idx % IPA_H;
    int bn = b * IPA_N + n;
    float* e = ext + ((size_t)((b * IPA_H + h) * IPA_N + n)) * IPA_KE;
    const float* s = sc + (size_t)bn * IPA_C + h * IPA_CH;
    #pragma unroll
    for (int c = 0; c < IPA_CH; ++c) e[c] = s[c];
    const float* g = pg + (size_t)bn * (IPA_H * IPA_P * 3) + h * (IPA_P * 3);
    float s2 = 0.0f;
    #pragma unroll
    for (int d = 0; d < 12; ++d) { float x = g[d]; e[32 + d] = x; s2 += x * x; }
    #pragma unroll
    for (int d = 44; d < IPA_KE; ++d) e[d] = 0.0f;
    sq[bn * IPA_H + h] = s2;
}

__global__ void ipa_build_k(const float* __restrict__ sc, const float* __restrict__ pg,
                            _Float16* __restrict__ ext, float* __restrict__ sq)
{
    int idx = blockIdx.x * blockDim.x + threadIdx.x;
    if (idx >= IPA_B * IPA_N * IPA_H) return;
    int b = idx / (IPA_N * IPA_H);
    int n = (idx / IPA_H) % IPA_N;
    int h = idx % IPA_H;
    int bn = b * IPA_N + n;
    _Float16* e = ext + ((size_t)((b * IPA_H + h) * IPA_N + n)) * IPA_KE;
    const float* s = sc + (size_t)bn * IPA_C + h * IPA_CH;
    #pragma unroll
    for (int c = 0; c < IPA_CH; ++c) e[c] = (_Float16)s[c];
    const float* g = pg + (size_t)bn * (IPA_H * IPA_P * 3) + h * (IPA_P * 3);
    float s2 = 0.0f;
    #pragma unroll
    for (int d = 0; d < 12; ++d) { float x = g[d]; e[32 + d] = (_Float16)x; s2 += x * x; }
    #pragma unroll
    for (int d = 44; d < IPA_KE; ++d) e[d] = (_Float16)0.0f;
    sq[bn * IPA_H + h] = s2;
}

__global__ void ipa_build_vT(const float* __restrict__ sc, const float* __restrict__ pg,
                             _Float16* __restrict__ extT)
{
    int idx = blockIdx.x * blockDim.x + threadIdx.x;
    if (idx >= IPA_B * IPA_N * IPA_H) return;
    int b = idx / (IPA_N * IPA_H);
    int m = (idx / IPA_H) % IPA_N;
    int h = idx % IPA_H;
    int bn = b * IPA_N + m;
    _Float16* e = extT + (size_t)(b * IPA_H + h) * IPA_KE * IPA_N + m;  // stride IPA_N per comp
    const float* s = sc + (size_t)bn * IPA_C + h * IPA_CH;
    #pragma unroll
    for (int c = 0; c < IPA_CH; ++c) e[(size_t)c * IPA_N] = (_Float16)s[c];
    const float* g = pg + (size_t)bn * (IPA_H * IPA_P * 3) + h * (IPA_P * 3);
    #pragma unroll
    for (int d = 0; d < 12; ++d) e[(size_t)(32 + d) * IPA_N] = (_Float16)g[d];
    #pragma unroll
    for (int d = 44; d < IPA_KE; ++d) e[(size_t)d * IPA_N] = (_Float16)0.0f;
}

// ---------------------------------------------------------------------------
// logits (= SCALE*qe.ke in attn buf) + pair_bias - 0.5*SCALE*(q2+k2), mask,
// softmax over m.  One block per (b,h,n) row; in place.  pb is [B*N*N, 16].
// ---------------------------------------------------------------------------
__global__ __launch_bounds__(256)
void ipa_bias_mask_softmax(float* __restrict__ attn,
                           const float* __restrict__ pb,
                           const float* __restrict__ q2,
                           const float* __restrict__ k2,
                           const float* __restrict__ mask)
{
    __shared__ float red[256];
    int row = blockIdx.x;                 // (b*H + h)*N + n
    int b = row / (IPA_H * IPA_N);
    int h = (row / IPA_N) % IPA_H;
    int n = row % IPA_N;
    float* L = attn + (size_t)row * IPA_N;
    float q2v = q2[(b * IPA_N + n) * IPA_H + h];
    float mn  = mask[b * IPA_N + n];

    float vals[2];
    float lmax = -3.0e38f;
    #pragma unroll
    for (int j = 0; j < 2; ++j) {
        int m = threadIdx.x + j * 256;
        float k2v = k2[(b * IPA_N + m) * IPA_H + h];
        size_t pidx = ((size_t)(b * IPA_N + n) * IPA_N + m) * 16 + h;
        float val = L[m] + pb[pidx] - 0.5f * IPA_SCALE * (q2v + k2v);
        if (mn * mask[b * IPA_N + m] == 0.0f) val = -1e9f;
        vals[j] = val;
        lmax = fmaxf(lmax, val);
    }
    red[threadIdx.x] = lmax;
    __syncthreads();
    for (int s = 128; s > 0; s >>= 1) {
        if (threadIdx.x < s) red[threadIdx.x] = fmaxf(red[threadIdx.x], red[threadIdx.x + s]);
        __syncthreads();
    }
    float rmax = red[0];
    __syncthreads();
    float lsum = 0.0f;
    #pragma unroll
    for (int j = 0; j < 2; ++j) { vals[j] = __expf(vals[j] - rmax); lsum += vals[j]; }
    red[threadIdx.x] = lsum;
    __syncthreads();
    for (int s = 128; s > 0; s >>= 1) {
        if (threadIdx.x < s) red[threadIdx.x] += red[threadIdx.x + s];
        __syncthreads();
    }
    float inv = 1.0f / red[0];
    #pragma unroll
    for (int j = 0; j < 2; ++j) L[threadIdx.x + j * 256] = vals[j] * inv;
}

// ---------------------------------------------------------------------------
// Split we[b,h,n,64] -> weighted_scalar[b,n,C], weighted_points[b,n,160] (padded)
// ---------------------------------------------------------------------------
__global__ void ipa_extract_weighted(const float* __restrict__ we,
                                     float* __restrict__ wsc,
                                     float* __restrict__ wpt)
{
    int idx = blockIdx.x * blockDim.x + threadIdx.x;
    if (idx >= IPA_B * IPA_N * IPA_H) return;
    int b = idx / (IPA_N * IPA_H);
    int n = (idx / IPA_H) % IPA_N;
    int h = idx % IPA_H;
    int bn = b * IPA_N + n;
    const float* e = we + ((size_t)((b * IPA_H + h) * IPA_N + n)) * IPA_KE;
    float* s = wsc + (size_t)bn * IPA_C + h * IPA_CH;
    #pragma unroll
    for (int c = 0; c < IPA_CH; ++c) s[c] = e[c];
    float* p = wpt + (size_t)bn * IPA_KPO + h * 12;
    #pragma unroll
    for (int d = 0; d < 12; ++d) p[d] = e[32 + d];
    if (h == 0) {                       // zero the K padding 144..159
        float* z = wpt + (size_t)bn * IPA_KPO + 144;
        #pragma unroll
        for (int d = 0; d < 16; ++d) z[d] = 0.0f;
    }
}

// ---------------------------------------------------------------------------
// out = LayerNorm(single + y) * g + b    (one block of 128 per (b,n))
// ---------------------------------------------------------------------------
__global__ __launch_bounds__(128)
void ipa_residual_layernorm(const float* __restrict__ single,
                            const float* __restrict__ y,
                            const float* __restrict__ g,
                            const float* __restrict__ bta,
                            float* __restrict__ out)
{
    __shared__ float red[128];
    int bn = blockIdx.x;
    const float* xs = single + (size_t)bn * IPA_C;
    const float* ys = y + (size_t)bn * IPA_C;
    float x[3];
    float lsum = 0.0f;
    #pragma unroll
    for (int j = 0; j < 3; ++j) {
        int c = threadIdx.x + j * 128;
        x[j] = xs[c] + ys[c];
        lsum += x[j];
    }
    red[threadIdx.x] = lsum;
    __syncthreads();
    for (int s = 64; s > 0; s >>= 1) {
        if (threadIdx.x < s) red[threadIdx.x] += red[threadIdx.x + s];
        __syncthreads();
    }
    float mu = red[0] * (1.0f / IPA_C);
    __syncthreads();
    float lv = 0.0f;
    #pragma unroll
    for (int j = 0; j < 3; ++j) { float d = x[j] - mu; lv += d * d; }
    red[threadIdx.x] = lv;
    __syncthreads();
    for (int s = 64; s > 0; s >>= 1) {
        if (threadIdx.x < s) red[threadIdx.x] += red[threadIdx.x + s];
        __syncthreads();
    }
    float r = rsqrtf(red[0] * (1.0f / IPA_C) + IPA_EPS);
    #pragma unroll
    for (int j = 0; j < 3; ++j) {
        int c = threadIdx.x + j * 128;
        out[(size_t)bn * IPA_C + c] = (x[j] - mu) * r * g[c] + bta[c];
    }
}

// ---------------------------------------------------------------------------
extern "C" void kernel_launch(void* const* d_in, const int* in_sizes, int n_in,
                              void* d_out, int out_size, void* d_ws, size_t ws_size,
                              hipStream_t stream) {
    const float* single = (const float*)d_in[0];
    const float* pair   = (const float*)d_in[1];
    const float* rot    = (const float*)d_in[2];
    const float* trans  = (const float*)d_in[3];
    const float* mask   = (const float*)d_in[4];
    const float* Wq  = (const float*)d_in[5];  const float* bq  = (const float*)d_in[6];
    const float* Wk  = (const float*)d_in[7];  const float* bk  = (const float*)d_in[8];
    const float* Wv  = (const float*)d_in[9];  const float* bv  = (const float*)d_in[10];
    const float* Wpb = (const float*)d_in[11]; const float* bpb = (const float*)d_in[12];
    const float* Wqp = (const float*)d_in[13]; const float* bqp = (const float*)d_in[14];
    const float* Wkp = (const float*)d_in[15]; const float* bkp = (const float*)d_in[16];
    const float* Wvp = (const float*)d_in[17]; const float* bvp = (const float*)d_in[18];
    const float* Wo  = (const float*)d_in[19]; const float* bo  = (const float*)d_in[20];
    const float* Wpo = (const float*)d_in[21]; const float* bpo = (const float*)d_in[22];
    const float* ln_g = (const float*)d_in[23]; const float* ln_b = (const float*)d_in[24];
    float* out = (float*)d_out;
    (void)ws_size; (void)in_sizes; (void)n_in; (void)out_size;

    // -------- workspace layout --------
    const size_t BN  = (size_t)IPA_B * IPA_N;               // 1024
    const size_t BHN = (size_t)IPA_B * IPA_H * IPA_N;       // 12288
    float* w = (float*)d_ws;
    size_t off = 0;
    auto falloc = [&](size_t n) { float* p = w + off; off += n; return p; };
    float* q    = falloc(BN * IPA_C);
    float* k    = falloc(BN * IPA_C);
    float* v    = falloc(BN * IPA_C);
    float* qp   = falloc(BN * 144);
    float* kp   = falloc(BN * 144);
    float* vp   = falloc(BN * 144);
    float* q2   = falloc(BN * IPA_H);
    float* k2   = falloc(BN * IPA_H);
    float* qe   = falloc(BHN * IPA_KE);
    float* we   = falloc(BHN * IPA_KE);
    float* pb   = falloc((size_t)IPA_B * IPA_N * IPA_N * 16);   // [B*N*N, 16]
    float* att  = falloc((size_t)IPA_B * IPA_H * IPA_N * IPA_N);
    float* wsc  = falloc(BN * IPA_C);
    float* wpt  = falloc(BN * IPA_KPO);
    float* sb   = falloc(BN * IPA_C);
    float* op   = falloc(BN * IPA_C);
    float* bpad = falloc(16);
    _Float16* hbase = (_Float16*)(w + off);
    size_t hoff = 0;
    auto halloc = [&](size_t n) { _Float16* p = hbase + hoff; hoff += n; return p; };
    _Float16* WqT  = halloc((size_t)384 * 384);
    _Float16* WkT  = halloc((size_t)384 * 384);
    _Float16* WvT  = halloc((size_t)384 * 384);
    _Float16* WqpT = halloc((size_t)144 * 384);
    _Float16* WkpT = halloc((size_t)144 * 384);
    _Float16* WvpT = halloc((size_t)144 * 384);
    _Float16* WpbT = halloc((size_t)16 * 128);
    _Float16* WpoT = halloc((size_t)384 * IPA_KPO);
    _Float16* WoT  = halloc((size_t)384 * 384);
    _Float16* keH  = halloc(BHN * IPA_KE);
    _Float16* veT  = halloc((size_t)IPA_B * IPA_H * IPA_KE * IPA_N);

    auto pack = [&](const float* W, const float* bsrc, _Float16* Wt, float* bp,
                    int K, int N, int Kpad, int Npad) {
        int tot = Kpad * Npad;
        ipa_pack_bt<<<(tot + 255) / 256, 256, 0, stream>>>(W, bsrc, Wt, bp, K, N, Kpad, Npad);
    };
    auto gemm = [&](const float* A, const _Float16* Bt, const float* bias,
                    const float* add, float* Cd,
                    int M, int Nn, int K, int lda, int ldb, int ldc,
                    long long sA, long long sB, long long sC,
                    int batch, float alpha) {
        int tiles = (M >> 4) * (Nn >> 4);
        dim3 grid((tiles + 7) / 8, batch);
        ipa_gemm_wmma<<<grid, 256, 0, stream>>>(A, Bt, bias, add, Cd,
                                                M, Nn, K, lda, ldb, ldc,
                                                sA, sB, sC, alpha);
    };

    // 0) pack weights to f16 column-major (k-contiguous) with padding
    pack(Wq,  nullptr, WqT,  nullptr, 384, 384, 384, 384);
    pack(Wk,  nullptr, WkT,  nullptr, 384, 384, 384, 384);
    pack(Wv,  nullptr, WvT,  nullptr, 384, 384, 384, 384);
    pack(Wqp, nullptr, WqpT, nullptr, 384, 144, 384, 144);
    pack(Wkp, nullptr, WkpT, nullptr, 384, 144, 384, 144);
    pack(Wvp, nullptr, WvpT, nullptr, 384, 144, 384, 144);
    pack(Wpb, bpb,     WpbT, bpad,    128, 12, 128, 16);
    pack(Wpo, nullptr, WpoT, nullptr, 144, 384, IPA_KPO, 384);
    pack(Wo,  nullptr, WoT,  nullptr, 384, 384, 384, 384);

    // 1) scalar + point projections of `single`
    gemm(single, WqT,  bq,  nullptr, q,  1024, 384, 384, 384, 384, 384, 0, 0, 0, 1, 1.0f);
    gemm(single, WkT,  bk,  nullptr, k,  1024, 384, 384, 384, 384, 384, 0, 0, 0, 1, 1.0f);
    gemm(single, WvT,  bv,  nullptr, v,  1024, 384, 384, 384, 384, 384, 0, 0, 0, 1, 1.0f);
    gemm(single, WqpT, bqp, nullptr, qp, 1024, 144, 384, 384, 384, 144, 0, 0, 0, 1, 1.0f);
    gemm(single, WkpT, bkp, nullptr, kp, 1024, 144, 384, 384, 384, 144, 0, 0, 0, 1, 1.0f);
    gemm(single, WvpT, bvp, nullptr, vp, 1024, 144, 384, 384, 384, 144, 0, 0, 0, 1, 1.0f);

    // 2) pair bias: the single pass over the 268 MB `pair` tensor (memory roof)
    gemm(pair, WpbT, bpad, nullptr, pb,
         IPA_B * IPA_N * IPA_N, 16, IPA_DP, IPA_DP, IPA_DP, 16, 0, 0, 0, 1, 1.0f);

    // 3) points -> global frames (in place)
    {
        int tot = IPA_B * IPA_N * IPA_H * IPA_P;
        int blks = (tot + 255) / 256;
        ipa_point_to_global<<<blks, 256, 0, stream>>>(qp, rot, trans);
        ipa_point_to_global<<<blks, 256, 0, stream>>>(kp, rot, trans);
        ipa_point_to_global<<<blks, 256, 0, stream>>>(vp, rot, trans);
    }

    // 4) extended rows (qe f32 A-side, ke f16 Bt-side, veT f16 Bt-side) + norms
    {
        int tot = IPA_B * IPA_N * IPA_H;
        int blks = (tot + 255) / 256;
        ipa_build_q <<<blks, 256, 0, stream>>>(q, qp, qe, q2);
        ipa_build_k <<<blks, 256, 0, stream>>>(k, kp, keH, k2);
        ipa_build_vT<<<blks, 256, 0, stream>>>(v, vp, veT);
    }

    // 5) fused scalar+point logits: att = SCALE * qe . ke^T   (batched over B*H)
    gemm(qe, keH, nullptr, nullptr, att,
         IPA_N, IPA_N, IPA_KE, IPA_KE, IPA_KE, IPA_N,
         (long long)IPA_N * IPA_KE, (long long)IPA_N * IPA_KE,
         (long long)IPA_N * IPA_N, IPA_B * IPA_H, IPA_SCALE);

    // 6) + pair_bias - 0.5*SCALE*(q2+k2), mask, softmax (in place)
    ipa_bias_mask_softmax<<<IPA_B * IPA_H * IPA_N, 256, 0, stream>>>(att, pb, q2, k2, mask);

    // 7) apply attention to fused values: we = att . ve   (batched over B*H)
    gemm(att, veT, nullptr, nullptr, we,
         IPA_N, IPA_KE, IPA_N, IPA_N, IPA_N, IPA_KE,
         (long long)IPA_N * IPA_N, (long long)IPA_KE * IPA_N,
         (long long)IPA_N * IPA_KE, IPA_B * IPA_H, 1.0f);

    // 8) split into weighted_scalar / weighted_points (padded to K=160)
    {
        int tot = IPA_B * IPA_N * IPA_H;
        int blks = (tot + 255) / 256;
        ipa_extract_weighted<<<blks, 256, 0, stream>>>(we, wsc, wpt);
    }

    // 9) sb = weighted_points @ Wpo + bpo + weighted_scalar
    gemm(wpt, WpoT, bpo, wsc, sb, 1024, 384, IPA_KPO, IPA_KPO, IPA_KPO, 384, 0, 0, 0, 1, 1.0f);

    // 10) op = sb @ Wo + bo
    gemm(sb, WoT, bo, nullptr, op, 1024, 384, 384, 384, 384, 384, 0, 0, 0, 1, 1.0f);

    // 11) out = LayerNorm(single + op)
    ipa_residual_layernorm<<<IPA_B * IPA_N, 128, 0, stream>>>(single, op, ln_g, ln_b, out);
}
